// MultiHeadAttention_89086211653703
// MI455X (gfx1250) — matrix operand order repaired and checked by reading, not compiled, not run on hardware
//
#include <hip/hip_runtime.h>
#include <hip/hip_bf16.h>

typedef __bf16 bf16;
typedef __attribute__((ext_vector_type(16))) __bf16 v16bf;
typedef __attribute__((ext_vector_type(8)))  __bf16 v8bf;
typedef __attribute__((ext_vector_type(8)))  float  v8f;
typedef __attribute__((ext_vector_type(4)))  int    v4i;

union FragA { v16bf v; v8bf h[2]; };

#define D_MODEL 1024
#define NH      16
#define NKV     4
#define DK      64
#define SEQ     2048
#define BATCH   2
#define QKV_N   (D_MODEL + 2 * NKV * DK)   // 1536
#define M_ROWS  (BATCH * SEQ)              // 4096
#define LOG2E   1.44269504088896340736f
#define SCL_Q   (0.125f * LOG2E)           // folded into Q at QKV epilogue

// -------- CDNA5 async global->LDS staging (probe-verified signature) --------
#if defined(__has_builtin)
#  if __has_builtin(__builtin_amdgcn_global_load_async_to_lds_b128) && \
      __has_builtin(__builtin_amdgcn_s_wait_asynccnt)
#    define HAVE_ASYNC_LDS 1
#  endif
#endif
#ifndef HAVE_ASYNC_LDS
#  define HAVE_ASYNC_LDS 0
#endif

#if HAVE_ASYNC_LDS
static __device__ __forceinline__ void async_b128(const bf16* g, bf16* l) {
    __builtin_amdgcn_global_load_async_to_lds_b128(
        (__attribute__((address_space(1))) v4i*)g,
        (__attribute__((address_space(3))) v4i*)l, 0, 0);
}
#endif

// -------- DPP16 butterfly reductions over 16-lane groups --------
template<int C>
static __device__ __forceinline__ float dpp_f(float x) {
    return __int_as_float(__builtin_amdgcn_update_dpp(
        __float_as_int(x), __float_as_int(x), C, 0xf, 0xf, true));
}
static __device__ __forceinline__ float redmax16(float x) {
    x = fmaxf(x, dpp_f<0x0B1>(x));   // quad_perm(1,0,3,2)  xor 1
    x = fmaxf(x, dpp_f<0x04E>(x));   // quad_perm(2,3,0,1)  xor 2
    x = fmaxf(x, dpp_f<0x141>(x));   // row_half_mirror     xor 7
    x = fmaxf(x, dpp_f<0x140>(x));   // row_mirror          xor 15
    return x;
}
static __device__ __forceinline__ float redsum16(float x) {
    x += dpp_f<0x0B1>(x);
    x += dpp_f<0x04E>(x);
    x += dpp_f<0x141>(x);
    x += dpp_f<0x140>(x);
    return x;
}

// ---------------------------------------------------------------- conversion
__global__ void cvt_f32_bf16(const float* __restrict__ src, bf16* __restrict__ dst, int n) {
    int i = (blockIdx.x * blockDim.x + threadIdx.x) * 8;
    int stride = gridDim.x * blockDim.x * 8;
    for (; i < n; i += stride) {
        float4 a = *(const float4*)(src + i);
        float4 b = *(const float4*)(src + i + 4);
        v8bf o;
        o[0] = (bf16)a.x; o[1] = (bf16)a.y; o[2] = (bf16)a.z; o[3] = (bf16)a.w;
        o[4] = (bf16)b.x; o[5] = (bf16)b.y; o[6] = (bf16)b.z; o[7] = (bf16)b.w;
        *(v8bf*)(dst + i) = o;
    }
}

// dst[n][k] = (bf16)src[k][n]  (one-time weight transpose, coalesced reads)
__global__ void cvt_transpose_bf16(const float* __restrict__ src, bf16* __restrict__ dst,
                                   int K, int N) {
    int n = blockIdx.x * blockDim.x + threadIdx.x;
    int k = blockIdx.y;
    if (n < N) dst[(long)n * K + k] = (bf16)src[(long)k * N + n];
}

// ---------------------------------------------------------------- WMMA GEMM
// C[M,N] = A[M,K] (bf16) * Bt[N,K] (bf16, pre-transposed) + bias
// Block 128x128, 8 waves (4x2), wave tile 32x64, KT=64, double-buffered async.
// MODE 0: fp32 C.  MODE 1: scatter into Q(prescaled) / K / V^T bf16 buffers.
template<int MODE>
__global__ __launch_bounds__(256)
void gemm_bf16(const bf16* __restrict__ A, const bf16* __restrict__ Bt,
               const float* __restrict__ bias, float* __restrict__ C,
               bf16* __restrict__ Qb, bf16* __restrict__ Kb, bf16* __restrict__ Vtb,
               int M, int N, int K) {
    const int ASZ = 128 * 72;
    __shared__ __align__(16) bf16 As[2 * 128 * 72];  // [buf][m][k] stride 72
    __shared__ __align__(16) bf16 Bs[2 * 128 * 72];  // [buf][n][k] stride 72

    const int tid  = threadIdx.x;
    const int wid  = tid >> 5, lane = tid & 31;
    const int lh   = lane >> 4, lm = lane & 15;
    const int wm   = wid >> 1, wn = wid & 1;
    const int bm   = blockIdx.y * 128;
    const int bn   = blockIdx.x * 128;

    v8f acc[2][4];
#pragma unroll
    for (int i = 0; i < 2; i++)
#pragma unroll
        for (int j = 0; j < 4; j++)
#pragma unroll
            for (int e = 0; e < 8; e++) acc[i][j][e] = 0.f;

    const int srow = tid >> 1, shalf = (tid & 1) * 32;   // 128 rows x 64 k tiles

    auto stage = [&](int buf, int kt) {
        const bf16* ag = A  + (long)(bm + srow) * K + kt + shalf;
        const bf16* bg = Bt + (long)(bn + srow) * K + kt + shalf;
        bf16* al = &As[buf * ASZ + srow * 72 + shalf];
        bf16* bl = &Bs[buf * ASZ + srow * 72 + shalf];
#if HAVE_ASYNC_LDS
#pragma unroll
        for (int u = 0; u < 4; u++) {
            async_b128(ag + u * 8, al + u * 8);
            async_b128(bg + u * 8, bl + u * 8);
        }
#else
#pragma unroll
        for (int u = 0; u < 4; u++) {
            *(v8bf*)(al + u * 8) = *(const v8bf*)(ag + u * 8);
            *(v8bf*)(bl + u * 8) = *(const v8bf*)(bg + u * 8);
        }
#endif
    };

    const int nIt = K >> 6;
    stage(0, 0);
    for (int it = 0; it < nIt; ++it) {
        const int cur = it & 1;
        if (it + 1 < nIt) {
            stage(cur ^ 1, (it + 1) << 6);    // prefetch next tile
#if HAVE_ASYNC_LDS
            __builtin_amdgcn_s_wait_asynccnt(8);   // current tile's 8 loads done
#endif
        } else {
#if HAVE_ASYNC_LDS
            __builtin_amdgcn_s_wait_asynccnt(0);
#endif
        }
        __syncthreads();

#pragma unroll
        for (int c = 0; c < 2; c++) {           // two 32-wide K slices
            FragA a[2], b[4];
#pragma unroll
            for (int i = 0; i < 2; i++) {
                int r = wm * 32 + i * 16 + lm;
                a[i].h[0] = *(const v8bf*)&As[cur * ASZ + r * 72 + c * 32 + lh * 8];
                a[i].h[1] = *(const v8bf*)&As[cur * ASZ + r * 72 + c * 32 + 16 + lh * 8];
            }
#pragma unroll
            for (int j = 0; j < 4; j++) {
                int r = wn * 64 + j * 16 + lm;
                b[j].h[0] = *(const v8bf*)&Bs[cur * ASZ + r * 72 + c * 32 + lh * 8];
                b[j].h[1] = *(const v8bf*)&Bs[cur * ASZ + r * 72 + c * 32 + 16 + lh * 8];
            }
#pragma unroll
            for (int i = 0; i < 2; i++)
#pragma unroll
                for (int j = 0; j < 4; j++)
                    acc[i][j] = __builtin_amdgcn_wmma_f32_16x16x32_bf16(
                        false, a[i].v, false, b[j].v, (short)0, acc[i][j], false, false);
        }
        __syncthreads();
    }

#pragma unroll
    for (int i = 0; i < 2; i++)
#pragma unroll
        for (int j = 0; j < 4; j++)
#pragma unroll
            for (int v = 0; v < 8; v++) {
                int row = bm + wm * 32 + i * 16 + v + 8 * lh;
                int col = bn + wn * 64 + j * 16 + lm;
                float val = acc[i][j][v] + bias[col];
                if (MODE == 0) {
                    C[(long)row * N + col] = val;
                } else {
                    int b_ = row >> 11, s = row & (SEQ - 1);
                    if (col < D_MODEL) {
                        int h = col >> 6, d = col & 63;
                        // fold softmax scale + log2e into Q
                        Qb[(((long)(b_ * NH + h) * SEQ) + s) * DK + d] = (bf16)(val * SCL_Q);
                    } else if (col < D_MODEL + NKV * DK) {
                        int c2 = col - D_MODEL, g = c2 >> 6, d = c2 & 63;
                        Kb[(((long)(b_ * NKV + g) * SEQ) + s) * DK + d] = (bf16)val;
                    } else {
                        int c2 = col - D_MODEL - NKV * DK, g = c2 >> 6, d = c2 & 63;
                        Vtb[(((long)(b_ * NKV + g) * DK) + d) * SEQ + s] = (bf16)val;  // V^T
                    }
                }
            }
}

// ---------------------------------------------------------------- flash attention
// grid: B*H*(S/128) blocks of 256 threads; wave w owns 16 query rows; KT=64 keys,
// double-buffered async K/V staging; ALiBi+mask folded into WMMA C-init.
__global__ __launch_bounds__(256)
void flash_attn(const bf16* __restrict__ Qb, const bf16* __restrict__ Kb,
                const bf16* __restrict__ Vtb, const int* __restrict__ mask,
                bf16* __restrict__ ctx) {
    const int KSZ = 64 * 72;
    __shared__ __align__(16) bf16 Ks[2 * 64 * 72];    // [buf][s_k][d]
    __shared__ __align__(16) bf16 Vs[2 * 64 * 72];    // [buf][d][s_k]
    __shared__ __align__(16) bf16 Ps[8 * 16 * 72];    // per-wave P tile [q][kk]

    int bx  = blockIdx.x;
    int qt  = bx & 15;
    int tmp = bx >> 4;
    int h   = tmp & 15;
    int b   = tmp >> 4;
    int hkv = h >> 2;
    float sl2 = exp2f(-(float)(h + 1)) * LOG2E;   // ALiBi slope in log2 domain

    int tid = threadIdx.x;
    int wid = tid >> 5, lane = tid & 31, lh = lane >> 4, lm = lane & 15;
    int q0  = qt * 128 + wid * 16;

    FragA aq[2];
    const bf16* qbase = Qb + ((long)(b * NH + h) * SEQ + q0 + lm) * DK;
#pragma unroll
    for (int dh = 0; dh < 2; dh++) {
        aq[dh].h[0] = *(const v8bf*)(qbase + dh * 32 + lh * 8);
        aq[dh].h[1] = *(const v8bf*)(qbase + dh * 32 + 16 + lh * 8);
    }

    // loop-invariant ALiBi terms (no int->float cvt inside the k loop)
    float qb[8];
#pragma unroll
    for (int v = 0; v < 8; v++) qb[v] = sl2 * (float)(q0 + v + 8 * lh);
    float kj[4];
#pragma unroll
    for (int j = 0; j < 4; j++) kj[j] = sl2 * (float)(j * 16 + lm);
    float ktf = 0.f, kstep = sl2 * 64.f;

    float mrow[8], lrow[8];
    v8f oacc[4];
#pragma unroll
    for (int v = 0; v < 8; v++) { mrow[v] = -3.0e38f; lrow[v] = 0.f; }
#pragma unroll
    for (int t = 0; t < 4; t++)
#pragma unroll
        for (int v = 0; v < 8; v++) oacc[t][v] = 0.f;

    const bf16* kbase = Kb  + (long)(b * NKV + hkv) * SEQ * DK;
    const bf16* vbase = Vtb + (long)(b * NKV + hkv) * DK * SEQ;
    const int strow = tid >> 2, stc = (tid & 3) * 16;   // 64 rows x 64 cols tiles

    auto stageKV = [&](int buf, int kt) {
        const bf16* kg = kbase + (long)(kt + strow) * DK + stc;
        const bf16* vg = vbase + (long)strow * SEQ + kt + stc;
        bf16* kl = &Ks[buf * KSZ + strow * 72 + stc];
        bf16* vl = &Vs[buf * KSZ + strow * 72 + stc];
#if HAVE_ASYNC_LDS
        async_b128(kg, kl);     async_b128(kg + 8, kl + 8);
        async_b128(vg, vl);     async_b128(vg + 8, vl + 8);
#else
        *(v8bf*)(kl)     = *(const v8bf*)(kg);
        *(v8bf*)(kl + 8) = *(const v8bf*)(kg + 8);
        *(v8bf*)(vl)     = *(const v8bf*)(vg);
        *(v8bf*)(vl + 8) = *(const v8bf*)(vg + 8);
#endif
    };

    const int nIt = SEQ >> 6;
    stageKV(0, 0);
    for (int it = 0; it < nIt; ++it) {
        const int kt = it << 6;
        const int cur = it & 1;
        if (it + 1 < nIt) {
            stageKV(cur ^ 1, kt + 64);         // prefetch next K/V tile
#if HAVE_ASYNC_LDS
            __builtin_amdgcn_s_wait_asynccnt(4);    // current tile's 4 loads done
#endif
        } else {
#if HAVE_ASYNC_LDS
            __builtin_amdgcn_s_wait_asynccnt(0);
#endif
        }
        __syncthreads();

        // bias-initialized score accumulators: sl2*q - sl2*k - 1e9*(mask==0)
        int mk[4];
#pragma unroll
        for (int j = 0; j < 4; j++) mk[j] = mask[b * SEQ + kt + j * 16 + lm];
        float kofs[4];
#pragma unroll
        for (int j = 0; j < 4; j++)
            kofs[j] = ktf + kj[j] + (mk[j] == 0 ? 1e9f : 0.f);
        v8f sacc[4];
#pragma unroll
        for (int j = 0; j < 4; j++)
#pragma unroll
            for (int v = 0; v < 8; v++) sacc[j][v] = qb[v] - kofs[j];

        // S = Q' K^T + bias : 16 q x 64 k per wave (Q pre-scaled by 0.125*log2e)
#pragma unroll
        for (int dh = 0; dh < 2; dh++)
#pragma unroll
            for (int j = 0; j < 4; j++) {
                int sr = j * 16 + lm;
                FragA bk;
                bk.h[0] = *(const v8bf*)&Ks[cur * KSZ + sr * 72 + dh * 32 + lh * 8];
                bk.h[1] = *(const v8bf*)&Ks[cur * KSZ + sr * 72 + dh * 32 + 16 + lh * 8];
                sacc[j] = __builtin_amdgcn_wmma_f32_16x16x32_bf16(
                    false, aq[dh].v, false, bk.v, (short)0, sacc[j], false, false);
            }

        // online softmax in log2 domain; row = v + 8*(lane>>4), col = lane&15
#pragma unroll
        for (int v = 0; v < 8; v++) {
            float s0 = sacc[0][v], s1 = sacc[1][v], s2 = sacc[2][v], s3 = sacc[3][v];
            float rm   = redmax16(fmaxf(fmaxf(s0, s1), fmaxf(s2, s3)));
            float mnew = fmaxf(mrow[v], rm);
            float corr = exp2f(mrow[v] - mnew);
            float p0 = exp2f(s0 - mnew), p1 = exp2f(s1 - mnew);
            float p2 = exp2f(s2 - mnew), p3 = exp2f(s3 - mnew);
            lrow[v] = lrow[v] * corr + redsum16((p0 + p1) + (p2 + p3));
            mrow[v] = mnew;
#pragma unroll
            for (int t = 0; t < 4; t++) oacc[t][v] *= corr;
            int m = v + 8 * lh;
            Ps[wid * 1152 + m * 72 + lm]      = (bf16)p0;
            Ps[wid * 1152 + m * 72 + 16 + lm] = (bf16)p1;
            Ps[wid * 1152 + m * 72 + 32 + lm] = (bf16)p2;
            Ps[wid * 1152 + m * 72 + 48 + lm] = (bf16)p3;
        }

        // P (C layout) -> A fragments via wave-private LDS
        FragA ap[2];
#pragma unroll
        for (int c = 0; c < 2; c++) {
            ap[c].h[0] = *(const v8bf*)&Ps[wid * 1152 + lm * 72 + c * 32 + lh * 8];
            ap[c].h[1] = *(const v8bf*)&Ps[wid * 1152 + lm * 72 + c * 32 + 16 + lh * 8];
        }

        // O += P V : 16 q x 64 d
#pragma unroll
        for (int t = 0; t < 4; t++) {
            int dr = t * 16 + lm;
#pragma unroll
            for (int c = 0; c < 2; c++) {
                FragA bv;
                bv.h[0] = *(const v8bf*)&Vs[cur * KSZ + dr * 72 + c * 32 + lh * 8];
                bv.h[1] = *(const v8bf*)&Vs[cur * KSZ + dr * 72 + c * 32 + 16 + lh * 8];
                oacc[t] = __builtin_amdgcn_wmma_f32_16x16x32_bf16(
                    false, ap[c].v, false, bv.v, (short)0, oacc[t], false, false);
            }
        }
        ktf += kstep;
        __syncthreads();
    }

#pragma unroll
    for (int v = 0; v < 8; v++) {
        float inv = 1.0f / lrow[v];
        int q = q0 + v + 8 * lh;
        bf16* obase = ctx + ((long)(b * SEQ + q)) * D_MODEL + h * DK;
#pragma unroll
        for (int t = 0; t < 4; t++) obase[t * 16 + lm] = (bf16)(oacc[t][v] * inv);
    }
}

// ---------------------------------------------------------------- launch
extern "C" void kernel_launch(void* const* d_in, const int* in_sizes, int n_in,
                              void* d_out, int out_size, void* d_ws, size_t ws_size,
                              hipStream_t stream) {
    (void)in_sizes; (void)n_in; (void)out_size; (void)ws_size;
    const float* x    = (const float*)d_in[0];
    const int*   mask = (const int*)d_in[1];
    const float* Wc   = (const float*)d_in[2];
    const float* bc   = (const float*)d_in[3];
    const float* Wo   = (const float*)d_in[4];
    const float* bo   = (const float*)d_in[5];
    float* out = (float*)d_out;

    char* w = (char*)d_ws;
    auto alloc = [&](size_t bytes) { char* p = w; w += (bytes + 255) & ~size_t(255); return p; };
    bf16* xb   = (bf16*)alloc((size_t)M_ROWS * D_MODEL * 2);
    bf16* WcbT = (bf16*)alloc((size_t)QKV_N * D_MODEL * 2);   // [N][K]
    bf16* WobT = (bf16*)alloc((size_t)D_MODEL * D_MODEL * 2); // [N][K]
    bf16* Qb   = (bf16*)alloc((size_t)BATCH * NH * SEQ * DK * 2);
    bf16* Kb   = (bf16*)alloc((size_t)BATCH * NKV * SEQ * DK * 2);
    bf16* Vtb  = (bf16*)alloc((size_t)BATCH * NKV * DK * SEQ * 2);
    bf16* ctx  = (bf16*)alloc((size_t)M_ROWS * D_MODEL * 2);

    cvt_f32_bf16<<<1024, 256, 0, stream>>>(x, xb, M_ROWS * D_MODEL);
    {   // Wc: [1024][1536] -> WcbT [1536][1024]
        dim3 g(QKV_N / 256, D_MODEL);
        cvt_transpose_bf16<<<g, 256, 0, stream>>>(Wc, WcbT, D_MODEL, QKV_N);
    }
    {   // Wo: [1024][1024] -> WobT [1024][1024]
        dim3 g(D_MODEL / 256, D_MODEL);
        cvt_transpose_bf16<<<g, 256, 0, stream>>>(Wo, WobT, D_MODEL, D_MODEL);
    }

    {   // QKV projection, scatter into Q(prescaled) / K / V^T
        dim3 grid(QKV_N / 128, M_ROWS / 128);
        gemm_bf16<1><<<grid, 256, 0, stream>>>(xb, WcbT, bc, nullptr,
                                               Qb, Kb, Vtb, M_ROWS, QKV_N, D_MODEL);
    }

    flash_attn<<<BATCH * NH * (SEQ / 128), 256, 0, stream>>>(Qb, Kb, Vtb, mask, ctx);

    {   // output projection (fp32 result)
        dim3 grid(D_MODEL / 128, M_ROWS / 128);
        gemm_bf16<0><<<grid, 256, 0, stream>>>(ctx, WobT, bo, out,
                                               nullptr, nullptr, nullptr,
                                               M_ROWS, D_MODEL, D_MODEL);
    }
}
